// PredictingModule_35304631173414
// MI455X (gfx1250) — matrix operand order repaired and checked by reading, not compiled, not run on hardware
//
#include <hip/hip_runtime.h>
#include <hip/hip_bf16.h>

typedef __attribute__((ext_vector_type(16))) _Float16 v16h;
typedef __attribute__((ext_vector_type(8)))  _Float16 v8h;
typedef __attribute__((ext_vector_type(8)))  float    v8f;

#define DIMD   128   // embed dim
#define NH1    128   // layer-1 N
#define NH2    64    // layer-2 N

// LDS fragment banks (B operands pre-swizzled: frag f, lane l, elem i at
// ((f*32)+l)*16 + i; elem i of lane l is K = kBase + (l<16 ? 0 : 8) + (i<8 ? i : 8+i),
// N = nBase + (l&15)).
#define W1_FRAGS 64  // 8 kc * 8 nc
#define W2_FRAGS 16  // 4 kc * 4 nc
#define W3_FRAGS 2   // 2 kc * 1 nc (N padded to 16, only col 0 nonzero)

__global__ __launch_bounds__(256)
void edge_mlp_kernel(const float* __restrict__ h_user,
                     const float* __restrict__ h_item,
                     const int*   __restrict__ src_uv,
                     const int*   __restrict__ dst_uv,
                     const int*   __restrict__ src_vu,
                     const int*   __restrict__ dst_vu,
                     const float* __restrict__ W1, const float* __restrict__ b1,
                     const float* __restrict__ W2, const float* __restrict__ b2,
                     const float* __restrict__ W3, const float* __restrict__ b3,
                     float* __restrict__ out, int nEdges)
{
    __shared__ __align__(32) _Float16 sW1[W1_FRAGS * 512];
    __shared__ __align__(32) _Float16 sW2[W2_FRAGS * 512];
    __shared__ __align__(32) _Float16 sW3[W3_FRAGS * 512];
    __shared__ __align__(32) _Float16 sScr[8][2][2048];  // per-wave, per-M-tile 16x128 f16

    const int tid = threadIdx.x;

    // ---- build swizzled f16 weight fragments in LDS (once per block) ----
    for (int idx = tid; idx < W1_FRAGS * 512; idx += 256) {
        int f = idx >> 9, r = idx & 511, ln = r >> 4, i = r & 15;
        int kc = f >> 3, nc = f & 7;
        int k = kc * 32 + (ln < 16 ? 0 : 8) + (i < 8 ? i : 8 + i);
        int n = nc * 16 + (ln & 15);
        sW1[idx] = (_Float16)W1[k * NH1 + n];
    }
    for (int idx = tid; idx < W2_FRAGS * 512; idx += 256) {
        int f = idx >> 9, r = idx & 511, ln = r >> 4, i = r & 15;
        int kc = f >> 2, nc = f & 3;
        int k = kc * 32 + (ln < 16 ? 0 : 8) + (i < 8 ? i : 8 + i);
        int n = nc * 16 + (ln & 15);
        sW2[idx] = (_Float16)W2[k * NH2 + n];
    }
    for (int idx = tid; idx < W3_FRAGS * 512; idx += 256) {
        int f = idx >> 9, r = idx & 511, ln = r >> 4, i = r & 15;
        int k = f * 32 + (ln < 16 ? 0 : 8) + (i < 8 ? i : 8 + i);
        sW3[idx] = ((ln & 15) == 0) ? (_Float16)W3[k] : (_Float16)0.0f;
    }
    __syncthreads();

    const int wave = tid >> 5;
    const int lane = tid & 31;
    const int m    = lane & 15;     // row within a 16-edge tile
    const bool hiH = lane >= 16;
    const int klo  = hiH ? 8 : 0;   // A-layout K sub-offset for this half-wave
    const int rbase = hiH ? 8 : 0;  // C-layout row base for this half-wave

    const int jobsPerDir = nEdges >> 5;            // 32 edges per wave job
    int t = blockIdx.x * 8 + wave;
    if (t >= 2 * jobsPerDir) return;               // wave-uniform

    const int*   sIdx; const int* dIdx;
    const float* sEmb; const float* dEmb;
    float* op;
    if (t < jobsPerDir) {                          // ('user','buys','item')
        sIdx = src_uv; dIdx = dst_uv; sEmb = h_user; dEmb = h_item; op = out;
    } else {                                       // ('item','bought-by','user')
        t -= jobsPerDir;
        sIdx = src_vu; dIdx = dst_vu; sEmb = h_item; dEmb = h_user; op = out + nEdges;
    }
    const int ebase = t * 32;

    const float* pu[2]; const float* pi[2];
#pragma unroll
    for (int mt = 0; mt < 2; mt++) {
        int e = ebase + mt * 16 + m;
        pu[mt] = sEmb + (long)sIdx[e] * DIMD;      // first half of concat
        pi[mt] = dEmb + (long)dIdx[e] * DIMD;      // second half of concat
    }

    // biases for this lane's N column (same for both half-waves)
    float b1v[8], b2v[4];
#pragma unroll
    for (int nc = 0; nc < 8; nc++) b1v[nc] = b1[nc * 16 + m];
#pragma unroll
    for (int nc = 0; nc < 4; nc++) b2v[nc] = b2[nc * 16 + m];
    const float b3v = b3[0];

    // ---------------- Layer 1: 2x [16,256] x [256,128], bias in C ----------------
    v8f acc1[2][8];
#pragma unroll
    for (int mt = 0; mt < 2; mt++)
#pragma unroll
        for (int nc = 0; nc < 8; nc++) acc1[mt][nc] = b1v[nc];   // splat bias

#pragma unroll
    for (int kc = 0; kc < 8; kc++) {
        const int kb = (kc & 3) * 32 + klo;
        v16h a[2];
#pragma unroll
        for (int mt = 0; mt < 2; mt++) {
            const float* row = (kc < 4) ? pu[mt] : pi[mt];
            float4 x0 = *(const float4*)(row + kb);
            float4 x1 = *(const float4*)(row + kb + 4);
            float4 x2 = *(const float4*)(row + kb + 16);
            float4 x3 = *(const float4*)(row + kb + 20);
            float xs[16] = { x0.x, x0.y, x0.z, x0.w, x1.x, x1.y, x1.z, x1.w,
                             x2.x, x2.y, x2.z, x2.w, x3.x, x3.y, x3.z, x3.w };
#pragma unroll
            for (int i = 0; i < 16; i++) a[mt][i] = (_Float16)xs[i];
        }
#pragma unroll
        for (int nc = 0; nc < 8; nc++) {
            const v16h b = *(const v16h*)&sW1[((kc * 8 + nc) * 32 + lane) * 16];
#pragma unroll
            for (int mt = 0; mt < 2; mt++)
                acc1[mt][nc] = __builtin_amdgcn_wmma_f32_16x16x32_f16(
                    false, a[mt], false, b, (short)0, acc1[mt][nc], false, false);
        }
    }

    // ReLU, spill h1 (f16, row-major 16x128 per M-tile) to per-wave scratch
#pragma unroll
    for (int mt = 0; mt < 2; mt++) {
        _Float16* sc = sScr[wave][mt];
#pragma unroll
        for (int nc = 0; nc < 8; nc++)
#pragma unroll
            for (int r = 0; r < 8; r++) {
                float v = acc1[mt][nc][r];
                v = v > 0.0f ? v : 0.0f;
                sc[(r + rbase) * NH1 + nc * 16 + m] = (_Float16)v;
            }
    }

    // ---------------- Layer 2: 2x [16,128] x [128,64], bias in C ----------------
    v8f acc2[2][4];
#pragma unroll
    for (int mt = 0; mt < 2; mt++)
#pragma unroll
        for (int nc = 0; nc < 4; nc++) acc2[mt][nc] = b2v[nc];

#pragma unroll
    for (int kc = 0; kc < 4; kc++) {
        const int kb = kc * 32 + klo;
        v16h a[2];
#pragma unroll
        for (int mt = 0; mt < 2; mt++) {
            const _Float16* sc = sScr[wave][mt];
            v8h lo = *(const v8h*)&sc[m * NH1 + kb];
            v8h hi = *(const v8h*)&sc[m * NH1 + kb + 16];
#pragma unroll
            for (int i = 0; i < 8; i++) { a[mt][i] = lo[i]; a[mt][i + 8] = hi[i]; }
        }
#pragma unroll
        for (int nc = 0; nc < 4; nc++) {
            const v16h b = *(const v16h*)&sW2[((kc * 4 + nc) * 32 + lane) * 16];
#pragma unroll
            for (int mt = 0; mt < 2; mt++)
                acc2[mt][nc] = __builtin_amdgcn_wmma_f32_16x16x32_f16(
                    false, a[mt], false, b, (short)0, acc2[mt][nc], false, false);
        }
    }

    // ReLU, spill h2 (f16, row-major 16x64 per M-tile)
#pragma unroll
    for (int mt = 0; mt < 2; mt++) {
        _Float16* sc = sScr[wave][mt];
#pragma unroll
        for (int nc = 0; nc < 4; nc++)
#pragma unroll
            for (int r = 0; r < 8; r++) {
                float v = acc2[mt][nc][r];
                v = v > 0.0f ? v : 0.0f;
                sc[(r + rbase) * NH2 + nc * 16 + m] = (_Float16)v;
            }
    }

    // ---------------- Layer 3: 2x [16,64] x [64,1 padded to 16], bias in C ----------------
    v8f acc3[2];
#pragma unroll
    for (int mt = 0; mt < 2; mt++) acc3[mt] = b3v;

#pragma unroll
    for (int kc = 0; kc < 2; kc++) {
        const int kb = kc * 32 + klo;
        const v16h b = *(const v16h*)&sW3[(kc * 32 + lane) * 16];
#pragma unroll
        for (int mt = 0; mt < 2; mt++) {
            const _Float16* sc = sScr[wave][mt];
            v8h lo = *(const v8h*)&sc[m * NH2 + kb];
            v8h hi = *(const v8h*)&sc[m * NH2 + kb + 16];
            v16h a;
#pragma unroll
            for (int i = 0; i < 8; i++) { a[i] = lo[i]; a[i + 8] = hi[i]; }
            acc3[mt] = __builtin_amdgcn_wmma_f32_16x16x32_f16(
                false, a, false, b, (short)0, acc3[mt], false, false);
        }
    }

    // sigmoid + store: N==0 column lives in lanes 0 (rows 0-7) and 16 (rows 8-15)
    if (m == 0) {
#pragma unroll
        for (int mt = 0; mt < 2; mt++) {
            float res[8];
#pragma unroll
            for (int r = 0; r < 8; r++)
                res[r] = 1.0f / (1.0f + __expf(-acc3[mt][r]));
            float* dst = op + ebase + mt * 16 + rbase;
            *(float4*)(dst)     = make_float4(res[0], res[1], res[2], res[3]);
            *(float4*)(dst + 4) = make_float4(res[4], res[5], res[6], res[7]);
        }
    }
}

extern "C" void kernel_launch(void* const* d_in, const int* in_sizes, int n_in,
                              void* d_out, int out_size, void* d_ws, size_t ws_size,
                              hipStream_t stream)
{
    const float* h_user = (const float*)d_in[0];
    const float* h_item = (const float*)d_in[1];
    const int*   src_uv = (const int*)d_in[2];
    const int*   dst_uv = (const int*)d_in[3];
    const int*   src_vu = (const int*)d_in[4];
    const int*   dst_vu = (const int*)d_in[5];
    const float* W1 = (const float*)d_in[6];
    const float* b1 = (const float*)d_in[7];
    const float* W2 = (const float*)d_in[8];
    const float* b2 = (const float*)d_in[9];
    const float* W3 = (const float*)d_in[10];
    const float* b3 = (const float*)d_in[11];
    float* out = (float*)d_out;

    const int E = in_sizes[2];                 // 1,000,000 (divisible by 32)
    const int totalJobs = 2 * (E >> 5);        // 32-edge wave jobs, both edge types
    const int blocks = (totalJobs + 7) / 8;    // 8 waves per 256-thread block

    edge_mlp_kernel<<<blocks, 256, 0, stream>>>(
        h_user, h_item, src_uv, dst_uv, src_vu, dst_vu,
        W1, b1, W2, b2, W3, b3, out, E);
}
